// CameraAwareFeatureFusion_40458591928953
// MI455X (gfx1250) — compile-verified, matrix operand order checked by reading, not run on hardware
//
// CDNA5 (gfx1250 / MI455X) implementation of CameraAwareFeatureFusion forward.
// All heavy math runs through v_wmma_f32_16x16x32_f16 (f32 accum). All GEMM/flash
// operands are pre-converted to f16 (weights + inputs cast once; producer kernels
// write f16 mirrors), so tile staging is a raw strided 2D copy done by the
// Tensor Data Mover (tensor_load_to_lds + s_wait_tensorcnt) when the builtin is
// available, with a plain f16 copy fallback otherwise. MHA is a flash-style
// streaming-softmax kernel; cross-view attention (d=1024) is two-pass.
#include <hip/hip_runtime.h>
#include <cstdint>
#include <cstddef>

typedef __attribute__((ext_vector_type(16))) _Float16 v16h;
typedef __attribute__((ext_vector_type(8)))  float    v8f;

struct __attribute__((aligned(16))) U128 { unsigned int a, b, c, d; };
union F16x16 { v16h v; U128 q[2]; _Float16 h[16]; };
struct __attribute__((aligned(8))) H4 { _Float16 a, b, c, d; };

__device__ __forceinline__ H4 cvt4(float4 f) {
  H4 h; h.a = (_Float16)f.x; h.b = (_Float16)f.y; h.c = (_Float16)f.z; h.d = (_Float16)f.w; return h;
}
// Load a 16-bit A/B fragment (16x32 layout: lane = row%16, K = kbase..+7 and kbase+16..+23)
__device__ __forceinline__ v16h ldfrag(const _Float16* p) {
  F16x16 f;
  f.q[0] = *(const U128*)(p);
  f.q[1] = *(const U128*)(p + 16);
  return f.v;
}
__device__ __forceinline__ float gelu_exact(float x) {
  return 0.5f * x * (1.0f + erff(x * 0.70710678118654752f));
}

// ---------------------------------------------------------------------------
// Tensor Data Mover: 2D f16 tile load Global->LDS with LDS row padding.
// D# layout per CDNA5 ISA ch.8 (group0: count/lds_addr/global_addr/type=2;
// group1: data_size, pad ctrl, tensor dims, tile dims, dim0 stride).
// Guarded: falls back to plain f16 copies if the builtin is absent.
// ---------------------------------------------------------------------------
#if __has_builtin(__builtin_amdgcn_tensor_load_to_lds) && __has_builtin(__builtin_amdgcn_s_wait_tensorcnt)
#define HAVE_TDM 1
typedef unsigned int v4u __attribute__((ext_vector_type(4)));
typedef int v4i __attribute__((ext_vector_type(4)));
typedef int v8i __attribute__((ext_vector_type(8)));

__device__ __forceinline__ void tdm_load_2d_f16(
    const _Float16* gsrc,          // tile start in global memory
    unsigned lds_byte_off,         // LDS destination byte offset
    unsigned tileK, unsigned tileR,            // tile_dim0 (contig elems), tile_dim1 (rows)
    unsigned long long strideElems,            // tensor_dim0_stride (elements)
    unsigned dim0, unsigned dim1,              // tensor dims (OOB guard)
    unsigned padIntervalCode, unsigned padAmountCode)
{
  unsigned long long ga = (unsigned long long)(const void*)gsrc;
  v4u g0 = {
    1u,                                                   // count=1 (valid user D#)
    lds_byte_off,
    (unsigned)ga,                                         // global_addr[31:0]
    (((unsigned)(ga >> 32)) & 0x01FFFFFFu) | 0x80000000u  // addr[56:32] | type=2
  };
  unsigned g1_0 = (1u << 16)                              // data_size = 2 bytes
                | (1u << 20)                              // pad_enable
                | ((padIntervalCode & 7u) << 22)
                | ((padAmountCode & 127u) << 25);
  unsigned g1_1 = (dim0 & 0xFFFFu) << 16;                 // tensor_dim0[15:0]
  unsigned g1_2 = ((dim0 >> 16) & 0xFFFFu) | ((dim1 & 0xFFFFu) << 16);
  unsigned g1_3 = ((dim1 >> 16) & 0xFFFFu) | ((tileK & 0xFFFFu) << 16); // tile_dim0
  unsigned g1_4 = (tileR & 0xFFFFu);                      // tile_dim1 (tile_dim2 = 0)
  unsigned g1_5 = (unsigned)(strideElems & 0xFFFFFFFFull);
  unsigned g1_6 = (unsigned)((strideElems >> 32) & 0xFFFFull);
  v8i g1 = { (int)g1_0, (int)g1_1, (int)g1_2, (int)g1_3,
             (int)g1_4, (int)g1_5, (int)g1_6, 0 };
  v4i z4 = { 0, 0, 0, 0 };
#if __clang_major__ >= 23
  v8i z8 = { 0, 0, 0, 0, 0, 0, 0, 0 };
  __builtin_amdgcn_tensor_load_to_lds(g0, g1, z4, z4, z8, 0);
#else
  __builtin_amdgcn_tensor_load_to_lds(g0, g1, z4, z4, 0);
#endif
}
#endif

// ---------------------------------------------------------------------------
// Tiled WMMA GEMM on f16 operands:
//   D   (f32, optional) = act(A * op(B) + bias) + residual
//   D16 (f16, optional) = same value (mirror for downstream GEMM consumption)
//   bT==0: B stored [N][K] -> A*B^T ; bT==1: B stored [K][N] -> A*B
// Requires M%128==0, N%128==0, K%32==0 (true at every call site).
// Block = 256 threads (8 waves); block tile 128x128; wave tile 32x64 (2x4 WMMA).
// ---------------------------------------------------------------------------
#define GLD 40   // LDS row: 32 data + 8 pad halves (64B data + 16B pad)

__global__ __launch_bounds__(256)
void gemm_h16_kernel(float* __restrict__ D, _Float16* __restrict__ D16, int ldd,
                     const _Float16* __restrict__ A, int lda,
                     const _Float16* __restrict__ B, int ldb,
                     const float* __restrict__ bias,
                     const float* __restrict__ residual, int ldr,
                     int M, int N, int K, int act, int bT)
{
  __shared__ __attribute__((aligned(16))) _Float16 As[128][GLD];
  __shared__ __attribute__((aligned(16))) _Float16 Bs[128][GLD];

  const int tid  = threadIdx.x;
  const int lane = tid & 31;
  const int wave = tid >> 5;
  const int wM   = wave >> 1;
  const int wN   = wave & 1;
  const int bM   = blockIdx.y * 128;
  const int bN   = blockIdx.x * 128;
  const int l16  = lane & 15;
  const int hs   = lane >> 4;
  const int kb   = hs * 8;

  v8f acc[2][4];
  #pragma unroll
  for (int i = 0; i < 2; ++i)
    #pragma unroll
    for (int j = 0; j < 4; ++j)
      acc[i][j] = {};

  for (int k0 = 0; k0 < K; k0 += 32) {
#ifdef HAVE_TDM
    if (wave == 0) {   // TDM engine does all straight-copy staging (EXEC-independent)
      tdm_load_2d_f16(A + (size_t)bM * lda + k0, (unsigned)(size_t)&As[0][0],
                      32, 128, (unsigned long long)lda, (unsigned)K, (unsigned)M, 3, 3);
      if (bT == 0)
        tdm_load_2d_f16(B + (size_t)bN * ldb + k0, (unsigned)(size_t)&Bs[0][0],
                        32, 128, (unsigned long long)ldb, (unsigned)K, (unsigned)N, 3, 3);
    }
#else
    // fallback: plain f16 copy, 128 rows x 32 halves = 512 x 16B chunks
    #pragma unroll
    for (int it = 0; it < 2; ++it) {
      int chunk = tid + it * 256;
      int row = chunk >> 2;
      int c8  = (chunk & 3) * 8;
      *(U128*)&As[row][c8] = *(const U128*)(A + (size_t)(bM + row) * lda + k0 + c8);
    }
    if (bT == 0) {
      #pragma unroll
      for (int it = 0; it < 2; ++it) {
        int chunk = tid + it * 256;
        int row = chunk >> 2;
        int c8  = (chunk & 3) * 8;
        *(U128*)&Bs[row][c8] = *(const U128*)(B + (size_t)(bN + row) * ldb + k0 + c8);
      }
    }
    if (k0 + 32 < K)
      __builtin_prefetch(A + (size_t)(bM + (tid >> 3)) * lda + (k0 + 32), 0, 1);
#endif
    if (bT == 1) {  // transposed staging (B[K][N]) — TDM cannot transpose
      const int nn = (tid & 31) * 4;
      const int kk = tid >> 5;
      #pragma unroll
      for (int it = 0; it < 4; ++it) {
        int kr = kk + it * 8;
        H4 v = *(const H4*)(B + (size_t)(k0 + kr) * ldb + (bN + nn));
        Bs[nn + 0][kr] = v.a;
        Bs[nn + 1][kr] = v.b;
        Bs[nn + 2][kr] = v.c;
        Bs[nn + 3][kr] = v.d;
      }
    }
#ifdef HAVE_TDM
    if (wave == 0) __builtin_amdgcn_s_wait_tensorcnt(0);
#endif
    __syncthreads();

    v16h af[2], bf[4];
    #pragma unroll
    for (int mt = 0; mt < 2; ++mt)
      af[mt] = ldfrag(&As[wM * 32 + mt * 16 + l16][kb]);
    #pragma unroll
    for (int nt = 0; nt < 4; ++nt)
      bf[nt] = ldfrag(&Bs[wN * 64 + nt * 16 + l16][kb]);

    #pragma unroll
    for (int mt = 0; mt < 2; ++mt)
      #pragma unroll
      for (int nt = 0; nt < 4; ++nt)
        acc[mt][nt] = __builtin_amdgcn_wmma_f32_16x16x32_f16(
            false, af[mt], false, bf[nt], (short)0, acc[mt][nt], false, false);

    __syncthreads();
  }

  #pragma unroll
  for (int mt = 0; mt < 2; ++mt)
    #pragma unroll
    for (int nt = 0; nt < 4; ++nt)
      #pragma unroll
      for (int i = 0; i < 8; ++i) {
        int row = bM + wM * 32 + mt * 16 + i + hs * 8;
        int col = bN + wN * 64 + nt * 16 + l16;
        float v = acc[mt][nt][i];
        if (bias)     v += bias[col];
        if (act == 2) v  = gelu_exact(v);
        if (residual) v += residual[(size_t)row * ldr + col];
        if (D)   D[(size_t)row * ldd + col]   = v;
        if (D16) D16[(size_t)row * ldd + col] = (_Float16)v;
      }
}

// ---------------------------------------------------------------------------
// Flash MHA over f16 QKV: per (view*batch, head, 64-q-row block). d=128, T=1024.
// qkv16: [vb*T][3072] halves, Q/K/V at col 0/1024/2048. out16: [vb*T][1024] f16.
// Scale applied to S in f32 (Q loaded raw). Block = 128 threads (4 waves).
// ---------------------------------------------------------------------------
__global__ __launch_bounds__(128)
void flash_mha_kernel(const _Float16* __restrict__ qkv, _Float16* __restrict__ out,
                      int T, int ldq, float scale)
{
  __shared__ __attribute__((aligned(16))) _Float16 Ks[32][136];   // [key][d], 16B row pad
  __shared__ __attribute__((aligned(16))) _Float16 Vs[128][40];   // [d][key] (transposed)
  __shared__ __attribute__((aligned(16))) _Float16 Ps[4][16][32]; // per-wave P re-layout

  const int tid  = threadIdx.x;
  const int lane = tid & 31;
  const int wave = tid >> 5;
  const int l16  = lane & 15;
  const int hs   = lane >> 4;
  const int kb   = hs * 8;
  const int h    = blockIdx.y;
  const int vb   = blockIdx.z;
  const int qbase = blockIdx.x * 64 + wave * 16;

  const _Float16* Qp = qkv + (size_t)vb * T * ldq + h * 128;
  const _Float16* Kp = Qp + 1024;
  const _Float16* Vp = Qp + 2048;

  v16h qf[4];
  {
    const _Float16* qr = Qp + (size_t)(qbase + l16) * ldq;
    #pragma unroll
    for (int c = 0; c < 4; ++c)
      qf[c] = ldfrag(qr + c * 32 + kb);
  }

  v8f o[8];
  #pragma unroll
  for (int i = 0; i < 8; ++i) o[i] = {};
  float m_i[8], l_i[8];
  #pragma unroll
  for (int i = 0; i < 8; ++i) { m_i[i] = -1e30f; l_i[i] = 0.f; }

  const int skey = tid >> 2;            // 0..31
  const int sdc  = (tid & 3) * 32;      // 0,32,64,96

  for (int kt = 0; kt < T; kt += 32) {
    // ---- stage K (TDM or plain copy) and V (manual transpose) ----
#ifdef HAVE_TDM
    if (wave == 0)
      tdm_load_2d_f16(Kp + (size_t)kt * ldq, (unsigned)(size_t)&Ks[0][0],
                      128, 32, (unsigned long long)ldq, 128u, (unsigned)T, 5, 3);
#else
    #pragma unroll
    for (int it = 0; it < 4; ++it) {
      int chunk = tid + it * 128;       // 32 rows x 16 chunks of 16B
      int row = chunk >> 4;
      int c8  = (chunk & 15) * 8;
      *(U128*)&Ks[row][c8] = *(const U128*)(Kp + (size_t)(kt + row) * ldq + c8);
    }
#endif
    {
      const _Float16* vrow = Vp + (size_t)(kt + skey) * ldq + sdc;
      #pragma unroll
      for (int e = 0; e < 8; ++e) {
        H4 v = *(const H4*)(vrow + e * 4);
        int d0 = sdc + e * 4;
        Vs[d0 + 0][skey] = v.a;
        Vs[d0 + 1][skey] = v.b;
        Vs[d0 + 2][skey] = v.c;
        Vs[d0 + 3][skey] = v.d;
      }
    }
#ifdef HAVE_TDM
    if (wave == 0) __builtin_amdgcn_s_wait_tensorcnt(0);
#endif
    __syncthreads();

    // ---- S = Q K^T (two 16x16 tiles covering 32 keys) ----
    v8f s0 = {}, s1 = {};
    #pragma unroll
    for (int c = 0; c < 4; ++c) {
      v16h b0 = ldfrag(&Ks[l16][c * 32 + kb]);
      v16h b1 = ldfrag(&Ks[16 + l16][c * 32 + kb]);
      s0 = __builtin_amdgcn_wmma_f32_16x16x32_f16(false, qf[c], false, b0, (short)0, s0, false, false);
      s1 = __builtin_amdgcn_wmma_f32_16x16x32_f16(false, qf[c], false, b1, (short)0, s1, false, false);
    }

    // ---- streaming softmax update (row reduction across 16 lanes) ----
    float corr[8];
    #pragma unroll
    for (int i = 0; i < 8; ++i) {
      float a0 = s0[i] * scale;
      float a1 = s1[i] * scale;
      float mx = fmaxf(a0, a1);
      #pragma unroll
      for (int off = 8; off > 0; off >>= 1)
        mx = fmaxf(mx, __shfl_xor(mx, off, 32));
      mx = fmaxf(mx, m_i[i]);
      float c0 = __expf(m_i[i] - mx);
      float p0 = __expf(a0 - mx);
      float p1 = __expf(a1 - mx);
      float rs = p0 + p1;
      #pragma unroll
      for (int off = 8; off > 0; off >>= 1)
        rs += __shfl_xor(rs, off, 32);
      l_i[i] = l_i[i] * c0 + rs;
      m_i[i] = mx;
      corr[i] = c0;
      int prow = i + hs * 8;
      Ps[wave][prow][l16]      = (_Float16)p0;
      Ps[wave][prow][l16 + 16] = (_Float16)p1;
    }
    #pragma unroll
    for (int nt = 0; nt < 8; ++nt)
      #pragma unroll
      for (int i = 0; i < 8; ++i)
        o[nt][i] *= corr[i];

    // ---- O += P * V (P re-read from wave-private LDS as A-fragment) ----
    v16h pf = ldfrag(&Ps[wave][l16][kb]);
    #pragma unroll
    for (int nt = 0; nt < 8; ++nt) {
      v16h vf = ldfrag(&Vs[nt * 16 + l16][kb]);
      o[nt] = __builtin_amdgcn_wmma_f32_16x16x32_f16(false, pf, false, vf, (short)0, o[nt], false, false);
    }
    __syncthreads();
  }

  float inv[8];
  #pragma unroll
  for (int i = 0; i < 8; ++i) inv[i] = 1.0f / l_i[i];
  #pragma unroll
  for (int nt = 0; nt < 8; ++nt)
    #pragma unroll
    for (int i = 0; i < 8; ++i) {
      int row = qbase + i + hs * 8;
      out[((size_t)(vb * T + row)) * 1024 + h * 128 + nt * 16 + l16] =
          (_Float16)(o[nt][i] * inv[i]);
    }
}

// ---------------------------------------------------------------------------
// fp32 -> f16 cast (weights & inputs, once per launch). n % 4 == 0.
// ---------------------------------------------------------------------------
__global__ __launch_bounds__(256)
void cast_f16_kernel(const float* __restrict__ src, _Float16* __restrict__ dst, size_t n)
{
  size_t i = ((size_t)blockIdx.x * 256 + threadIdx.x) * 4;
  if (i >= n) return;
  float4 v = *(const float4*)(src + i);
  *(H4*)(dst + i) = cvt4(v);
}

// ---------------------------------------------------------------------------
// Tiny camera MLPs: one block per view, computes cam_emb[v][1024] and ce[v][1024].
// ---------------------------------------------------------------------------
__global__ __launch_bounds__(256)
void camera_kernel(const float* __restrict__ ext, const float* __restrict__ intr,
                   const float* rot1W, const float* rot1b, const float* rot2W, const float* rot2b,
                   const float* tr1W,  const float* tr1b,  const float* tr2W,  const float* tr2b,
                   const float* in1W,  const float* in1b,  const float* in2W,  const float* in2b,
                   const float* projW, const float* projb,
                   const float* cam1W, const float* cam1b, const float* cam2W, const float* cam2b,
                   float* __restrict__ cam_emb, float* __restrict__ ce)
{
  const int v = blockIdx.x, tid = threadIdx.x;
  __shared__ float vin[16];
  __shared__ float h1[1024];
  __shared__ float h2[1024];
  __shared__ float hc[512];
  const float* E = ext + v * 16;
  const float* Kk = intr + v * 9;
  if (tid == 0) {
    vin[0]=E[0];  vin[1]=E[1];  vin[2]=E[2];
    vin[3]=E[4];  vin[4]=E[5];  vin[5]=E[6];
    vin[6]=E[8];  vin[7]=E[9];  vin[8]=E[10];
    vin[9]=E[3];  vin[10]=E[7]; vin[11]=E[11];
    vin[12]=Kk[0]; vin[13]=Kk[4]; vin[14]=Kk[2]; vin[15]=Kk[5];
  }
  __syncthreads();
  for (int o = tid; o < 1024; o += 256) {
    float s;
    if (o < 512)      { s = rot1b[o];     for (int j = 0; j < 9; ++j) s += rot1W[o*9+j]   * vin[j];      }
    else if (o < 768) { int oo=o-512; s = tr1b[oo]; for (int j = 0; j < 3; ++j) s += tr1W[oo*3+j] * vin[9+j];  }
    else              { int oo=o-768; s = in1b[oo]; for (int j = 0; j < 4; ++j) s += in1W[oo*4+j] * vin[12+j]; }
    h1[o] = fmaxf(s, 0.f);
  }
  __syncthreads();
  for (int o = tid; o < 1024; o += 256) {
    float s;
    if (o < 512)      { s = rot2b[o];     for (int j = 0; j < 512; ++j) s += rot2W[o*512+j]   * h1[j];     }
    else if (o < 768) { int oo=o-512; s = tr2b[oo]; for (int j = 0; j < 256; ++j) s += tr2W[oo*256+j] * h1[512+j]; }
    else              { int oo=o-768; s = in2b[oo]; for (int j = 0; j < 256; ++j) s += in2W[oo*256+j] * h1[768+j]; }
    h2[o] = s;
  }
  __syncthreads();
  for (int o = tid; o < 1024; o += 256) {
    float s = projb[o];
    for (int j = 0; j < 1024; ++j) s += projW[o*1024+j] * h2[j];
    cam_emb[(size_t)v * 1024 + o] = s;
  }
  for (int o = tid; o < 512; o += 256) {
    float s = cam1b[o];
    for (int j = 0; j < 16; ++j) s += cam1W[o*16+j] * vin[j];
    hc[o] = fmaxf(s, 0.f);
  }
  __syncthreads();
  for (int o = tid; o < 1024; o += 256) {
    float s = cam2b[o];
    for (int j = 0; j < 512; ++j) s += cam2W[o*512+j] * hc[j];
    ce[(size_t)v * 1024 + o] = s;
  }
}

// ---------------------------------------------------------------------------
// LayerNorm: one block per row of H=1024; output is the f16 GEMM feed.
// ---------------------------------------------------------------------------
__global__ __launch_bounds__(256)
void layernorm_kernel(const float* __restrict__ x, const float* __restrict__ g,
                      const float* __restrict__ b, _Float16* __restrict__ y, int H)
{
  const int tid = threadIdx.x, lane = tid & 31, w = tid >> 5;
  const float* xr = x + (size_t)blockIdx.x * H;
  __shared__ float red[20];
  float s = 0.f, s2 = 0.f;
  for (int c = tid; c < H; c += 256) { float v = xr[c]; s += v; s2 += v * v; }
  #pragma unroll
  for (int o = 16; o > 0; o >>= 1) { s += __shfl_xor(s, o, 32); s2 += __shfl_xor(s2, o, 32); }
  if (lane == 0) { red[w] = s; red[8 + w] = s2; }
  __syncthreads();
  if (tid == 0) {
    float a = 0.f, c2 = 0.f;
    for (int i = 0; i < 8; ++i) { a += red[i]; c2 += red[8 + i]; }
    float mean = a / H;
    red[16] = mean;
    red[17] = rsqrtf(c2 / H - mean * mean + 1e-5f);
  }
  __syncthreads();
  const float mean = red[16], rstd = red[17];
  _Float16* yr = y + (size_t)blockIdx.x * H;
  for (int c = tid; c < H; c += 256)
    yr[c] = (_Float16)((xr[c] - mean) * rstd * g[c] + b[c]);
}

// ---------------------------------------------------------------------------
// Row softmax with scale; f32 scratch in place, f16 mirror out for the PV GEMM.
// ---------------------------------------------------------------------------
__global__ __launch_bounds__(256)
void softmax_rows_kernel(float* __restrict__ S, _Float16* __restrict__ S16,
                         int N, float scale)
{
  const int tid = threadIdx.x, lane = tid & 31, w = tid >> 5;
  float* r = S + (size_t)blockIdx.x * N;
  _Float16* r16 = S16 + (size_t)blockIdx.x * N;
  __shared__ float red[12];
  float mx = -1e30f;
  for (int c = tid; c < N; c += 256) mx = fmaxf(mx, r[c] * scale);
  #pragma unroll
  for (int o = 16; o > 0; o >>= 1) mx = fmaxf(mx, __shfl_xor(mx, o, 32));
  if (lane == 0) red[w] = mx;
  __syncthreads();
  if (tid == 0) { float m2 = red[0]; for (int i = 1; i < 8; ++i) m2 = fmaxf(m2, red[i]); red[8] = m2; }
  __syncthreads();
  mx = red[8];
  float sum = 0.f;
  for (int c = tid; c < N; c += 256) { float e = __expf(r[c] * scale - mx); r[c] = e; sum += e; }
  #pragma unroll
  for (int o = 16; o > 0; o >>= 1) sum += __shfl_xor(sum, o, 32);
  if (lane == 0) red[w] = sum;
  __syncthreads();
  if (tid == 0) { float s2 = 0.f; for (int i = 0; i < 8; ++i) s2 += red[i]; red[9] = 1.0f / s2; }
  __syncthreads();
  const float inv = red[9];
  for (int c = tid; c < N; c += 256) r16[c] = (_Float16)(r[c] * inv);
}

// ---------------------------------------------------------------------------
// y16[m][c] = (f16)( x[m][c] + vec[m / rowsPerVec][c] )
// ---------------------------------------------------------------------------
__global__ __launch_bounds__(256)
void add_vec_kernel(const float* __restrict__ x, const float* __restrict__ vec,
                    _Float16* __restrict__ y, int H, int rowsPerVec, size_t total)
{
  size_t idx = (size_t)blockIdx.x * 256 + threadIdx.x;
  if (idx >= total) return;
  int c   = (int)(idx % H);
  int row = (int)(idx / H);
  int vi  = row / rowsPerVec;
  y[idx] = (_Float16)(x[idx] + vec[(size_t)vi * H + c]);
}

// ---------------------------------------------------------------------------
// fused16[b,t,c] = (f16) mean_v( x[v,b,t,c] + 0.2*addb[v,b,t,c] )
// ---------------------------------------------------------------------------
__global__ __launch_bounds__(256)
void fuse_mean_kernel(const float* __restrict__ x, const float* __restrict__ addb,
                      _Float16* __restrict__ fused, size_t total)
{
  size_t idx = (size_t)blockIdx.x * 256 + threadIdx.x;
  if (idx >= total) return;
  int c     = (int)(idx & 1023);
  size_t bt = idx >> 10;
  int b     = (int)(bt >> 10);
  int t     = (int)(bt & 1023);
  float s = 0.f;
  #pragma unroll
  for (int v = 0; v < 4; ++v) {
    size_t m = (((size_t)(v * 2 + b) * 1024) + t) * 1024 + c;
    s += x[m] + 0.2f * addb[m];
  }
  fused[idx] = (_Float16)(s * 0.25f);
}

// ---------------------------------------------------------------------------
// Host orchestration
// ---------------------------------------------------------------------------
extern "C" void kernel_launch(void* const* d_in, const int* in_sizes, int n_in,
                              void* d_out, int out_size, void* d_ws, size_t ws_size,
                              hipStream_t stream)
{
  (void)in_sizes; (void)n_in; (void)out_size; (void)ws_size;
  // Input order: view_features, extrinsics, intrinsics, then params flattened
  // as a jax pytree (sorted dict keys): blocks[0..2]{ff1,ff2,in_proj,ln2,out_proj},
  // cam{in1,in2,proj,rot1,rot2,tr1,tr2}, corr{cam1,cam2,k,q}, feat_proj, out_proj.
  const float* vf   = (const float*)d_in[0];
  const float* ext  = (const float*)d_in[1];
  const float* intr = (const float*)d_in[2];

  const float *ff1W[3], *ff1b[3], *ff2W[3], *ff2b[3], *inW[3], *inb[3],
              *lng[3], *lnb[3], *opW[3], *opb[3];
  for (int L = 0; L < 3; ++L) {
    int b0 = 3 + L * 10;
    ff1W[L] = (const float*)d_in[b0 + 0]; ff1b[L] = (const float*)d_in[b0 + 1];
    ff2W[L] = (const float*)d_in[b0 + 2]; ff2b[L] = (const float*)d_in[b0 + 3];
    inW[L]  = (const float*)d_in[b0 + 4]; inb[L]  = (const float*)d_in[b0 + 5];
    lng[L]  = (const float*)d_in[b0 + 6]; lnb[L]  = (const float*)d_in[b0 + 7];
    opW[L]  = (const float*)d_in[b0 + 8]; opb[L]  = (const float*)d_in[b0 + 9];
  }
  const float* in1W  = (const float*)d_in[33]; const float* in1b  = (const float*)d_in[34];
  const float* in2W  = (const float*)d_in[35]; const float* in2b  = (const float*)d_in[36];
  const float* projW = (const float*)d_in[37]; const float* projb = (const float*)d_in[38];
  const float* rot1W = (const float*)d_in[39]; const float* rot1b = (const float*)d_in[40];
  const float* rot2W = (const float*)d_in[41]; const float* rot2b = (const float*)d_in[42];
  const float* tr1W  = (const float*)d_in[43]; const float* tr1b  = (const float*)d_in[44];
  const float* tr2W  = (const float*)d_in[45]; const float* tr2b  = (const float*)d_in[46];
  const float* cam1W = (const float*)d_in[47]; const float* cam1b = (const float*)d_in[48];
  const float* cam2W = (const float*)d_in[49]; const float* cam2b = (const float*)d_in[50];
  const float* kW    = (const float*)d_in[51]; const float* kb_   = (const float*)d_in[52];
  const float* qW    = (const float*)d_in[53]; const float* qb    = (const float*)d_in[54];
  const float* fpW   = (const float*)d_in[55]; const float* fpb   = (const float*)d_in[56];
  const float* outW  = (const float*)d_in[57]; const float* outb  = (const float*)d_in[58];

  // ---- workspace layout (bytes); peak ~370 MB ----
  char* base = (char*)d_ws;
  size_t P = 0;
  auto allocF = [&](size_t n) { float* p = (float*)(base + P); P += n * 4; return p; };
  auto allocH = [&](size_t n) { _Float16* p = (_Float16*)(base + P); P += n * 2; return p; };
  const size_t MH = (size_t)8192 * 1024;

  float* cam_emb = allocF(4096);
  float* ce      = allocF(4096);
  float* x       = allocF(MH);                 // f32 token state (residual chain)
  float* addb    = allocF(MH);                 // cross-view accumulator
  float* Sbuf    = allocF((size_t)1024 * 1024);

  _Float16* vf16   = allocH(MH);
  _Float16* x16    = allocH(MH);               // f16 mirror of x
  _Float16* qk16   = allocH(MH);               // x + cam_pos / feats + ce / (reused) fused
  _Float16* qkv16  = allocH(MH * 3);           // fused QKV; later qbuf/kbuf
  _Float16* hbuf16 = allocH(MH);               // LN output
  _Float16* ffh16  = allocH(MH * 4);           // FF hidden (f16 only)
  _Float16* attn16 = allocH(MH);               // MHA output
  _Float16* S16    = allocH((size_t)1024 * 1024);
  _Float16* fpW16  = allocH((size_t)1024 * 1024);
  _Float16* qW16   = allocH((size_t)1024 * 1024);
  _Float16* kW16   = allocH((size_t)1024 * 1024);
  _Float16* outW16 = allocH((size_t)1024 * 1024);
  _Float16 *inW16[3], *opW16[3], *ff1W16[3], *ff2W16[3];
  for (int L = 0; L < 3; ++L) {
    inW16[L]  = allocH((size_t)3072 * 1024);
    opW16[L]  = allocH((size_t)1024 * 1024);
    ff1W16[L] = allocH((size_t)4096 * 1024);
    ff2W16[L] = allocH((size_t)4096 * 1024);
  }
  _Float16* qbuf16  = qkv16;
  _Float16* kbuf16  = qkv16 + MH;
  _Float16* fused16 = qk16;

  auto cast = [&](const float* s, _Float16* d, size_t n) {
    cast_f16_kernel<<<(unsigned)((n / 4 + 255) / 256), 256, 0, stream>>>(s, d, n);
  };
  auto gemm = [&](float* Dp, _Float16* D16p, int ldd, const _Float16* Ap, int lda,
                  const _Float16* Bp, int ldb, const float* biasp,
                  const float* resp, int ldr, int M, int N, int K, int act, int bT) {
    dim3 g(N / 128, M / 128);
    gemm_h16_kernel<<<g, 256, 0, stream>>>(Dp, D16p, ldd, Ap, lda, Bp, ldb,
                                           biasp, resp, ldr, M, N, K, act, bT);
  };

  // 0) one-time f16 casts (weights + input features)
  cast(vf, vf16, MH);
  cast(fpW, fpW16, (size_t)1024 * 1024);
  cast(qW, qW16, (size_t)1024 * 1024);
  cast(kW, kW16, (size_t)1024 * 1024);
  cast(outW, outW16, (size_t)1024 * 1024);
  for (int L = 0; L < 3; ++L) {
    cast(inW[L],  inW16[L],  (size_t)3072 * 1024);
    cast(opW[L],  opW16[L],  (size_t)1024 * 1024);
    cast(ff1W[L], ff1W16[L], (size_t)4096 * 1024);
    cast(ff2W[L], ff2W16[L], (size_t)4096 * 1024);
  }

  // 1) camera embeddings + correlation camera vectors
  camera_kernel<<<dim3(4), 256, 0, stream>>>(ext, intr,
      rot1W, rot1b, rot2W, rot2b, tr1W, tr1b, tr2W, tr2b,
      in1W, in1b, in2W, in2b, projW, projb,
      cam1W, cam1b, cam2W, cam2b, cam_emb, ce);

  // 2) feat_proj
  gemm(x, x16, 1024, vf16, 1024, fpW16, 1024, fpb, nullptr, 0, 8192, 1024, 1024, 0, 0);

  const unsigned gridMH = (unsigned)((MH + 255) / 256);

  // 3) transformer blocks
  for (int L = 0; L < 3; ++L) {
    add_vec_kernel<<<gridMH, 256, 0, stream>>>(x, cam_emb, qk16, 1024, 2048, MH);
    gemm(nullptr, qkv16, 3072, qk16, 1024, inW16[L], 1024, inb[L], nullptr, 0,
         8192, 2048, 1024, 0, 0);
    gemm(nullptr, qkv16 + 2048, 3072, x16, 1024, inW16[L] + (size_t)2048 * 1024, 1024,
         inb[L] + 2048, nullptr, 0, 8192, 1024, 1024, 0, 0);
    flash_mha_kernel<<<dim3(16, 8, 8), 128, 0, stream>>>(qkv16, attn16, 1024, 3072,
                                                         0.08838834764831845f);
    gemm(x, x16, 1024, attn16, 1024, opW16[L], 1024, opb[L], x, 1024,
         8192, 1024, 1024, 0, 0);
    layernorm_kernel<<<8192, 256, 0, stream>>>(x, lng[L], lnb[L], hbuf16, 1024);
    gemm(nullptr, ffh16, 4096, hbuf16, 1024, ff1W16[L], 1024, ff1b[L], nullptr, 0,
         8192, 4096, 1024, 2, 0);
    gemm(x, x16, 1024, ffh16, 4096, ff2W16[L], 4096, ff2b[L], x, 1024,
         8192, 1024, 4096, 0, 0);
  }

  // 4) cross-view correlation q/k
  add_vec_kernel<<<gridMH, 256, 0, stream>>>(x, ce, qk16, 1024, 2048, MH);
  gemm(nullptr, qbuf16, 1024, qk16, 1024, qW16, 1024, qb, nullptr, 0, 8192, 1024, 1024, 0, 0);
  gemm(nullptr, kbuf16, 1024, qk16, 1024, kW16, 1024, kb_, nullptr, 0, 8192, 1024, 1024, 0, 0);

  // 5) pairwise cross-view attention (d=1024): two-pass through Sbuf
  const float cscale = 1.0f / 32.0f;   // 1/sqrt(1024)
  for (int i = 0; i < 4; ++i)
    for (int b = 0; b < 2; ++b) {
      float*           dsti = addb   + (size_t)(i * 2 + b) * 1024 * 1024;
      const _Float16*  qi   = qbuf16 + (size_t)(i * 2 + b) * 1024 * 1024;
      bool first = true;
      for (int j = 0; j < 4; ++j) {
        if (j == i) continue;
        const _Float16* kj = kbuf16 + (size_t)(j * 2 + b) * 1024 * 1024;
        const _Float16* fj = x16    + (size_t)(j * 2 + b) * 1024 * 1024;
        gemm(Sbuf, nullptr, 1024, qi, 1024, kj, 1024, nullptr, nullptr, 0,
             1024, 1024, 1024, 0, 0);
        softmax_rows_kernel<<<1024, 256, 0, stream>>>(Sbuf, S16, 1024, cscale);
        gemm(dsti, nullptr, 1024, S16, 1024, fj, 1024, nullptr,
             first ? nullptr : dsti, 1024, 1024, 1024, 1024, 0, 1);
        first = false;
      }
    }

  // 6) fuse views + final projection into d_out (fp32)
  const size_t totalBT = (size_t)2048 * 1024;
  fuse_mean_kernel<<<(unsigned)((totalBT + 255) / 256), 256, 0, stream>>>(x, addb, fused16, totalBT);
  gemm((float*)d_out, nullptr, 1024, fused16, 1024, outW16, 1024, outb, nullptr, 0,
       2048, 1024, 1024, 0, 0);
}
